// MANNet_28209345200803
// MI455X (gfx1250) — compile-verified, matrix operand order checked by reading, not run on hardware
//
#include <hip/hip_runtime.h>
#include <hip/hip_bf16.h>
#include <math.h>

// ---------------- model dims ----------------
#define BB 4
#define SS 256
#define DD 300
#define DP 320      // D padded to mult of 32
#define EE 128
#define E2 256
#define H3 384
#define E12 1536
#define LL 20

typedef _Float16 half8 __attribute__((ext_vector_type(8)));
typedef _Float16 v16h  __attribute__((ext_vector_type(16)));
typedef float    v8f   __attribute__((ext_vector_type(8)));

__device__ __forceinline__ float sigmf(float x) { return 1.f / (1.f + __expf(-x)); }

// ---------------- embedding gather ----------------
__global__ void embed_gather(const int* __restrict__ idx, const float* __restrict__ emb,
                             float* __restrict__ x) {
  int bs = blockIdx.x;
  long long row = (long long)idx[bs] * DD;
  for (int d = threadIdx.x; d < DD; d += blockDim.x)
    x[(long long)bs * DD + d] = emb[row + d];
}

// ---------------- f32 -> f16 with K padding AND row padding ----------------
// out is [R, Kp]; rows >= Rv and cols >= K are zero-filled so WMMA fragment
// loads never need bounds checks.
__global__ void cvt_f16_pad(const float* __restrict__ in, _Float16* __restrict__ out,
                            int Rv, int R, int K, int Kp) {
  long long i = (long long)blockIdx.x * blockDim.x + threadIdx.x;
  if (i >= (long long)R * Kp) return;
  int r = (int)(i / Kp), k = (int)(i % Kp);
  out[i] = (r < Rv && k < K) ? (_Float16)in[(long long)r * K + k] : (_Float16)0.f;
}

// ---------------- WMMA GEMM: C[M,N] = A[M,Kp] @ W[N,Kp]^T (+bias), batched ----------------
// one wave per 16x16 output tile; A/B fragments: two 16B runs per lane per K-step.
// A and W are row-padded to multiples of 16 rows -> unconditional inner loop.
__global__ void wmma_gemm(const _Float16* __restrict__ A, const _Float16* __restrict__ W,
                          const float* __restrict__ bias, float* __restrict__ C,
                          int M, int N, int Kp,
                          long long sA, long long sW, long long sC) {
  int zb = blockIdx.z;
  A += (long long)zb * sA; W += (long long)zb * sW; C += (long long)zb * sC;
  int nt = blockIdx.x, mt = blockIdx.y;
  int lane = threadIdx.x & 31;
  int hi = lane >> 4, l16 = lane & 15;
  int arow = mt * 16 + l16;
  int wrow = nt * 16 + l16;
  const _Float16* ap = A + (long long)arow * Kp;
  const _Float16* wp = W + (long long)wrow * Kp;
  int r1 = hi * 8, r2 = 16 + hi * 8;
  v8f c = {};
  for (int k = 0; k < Kp; k += 32) {
    half8 a1 = *(const half8*)(ap + k + r1);
    half8 a2 = *(const half8*)(ap + k + r2);
    half8 b1 = *(const half8*)(wp + k + r1);
    half8 b2 = *(const half8*)(wp + k + r2);
    v16h av, bv;
#pragma unroll
    for (int i = 0; i < 8; ++i) { av[i] = a1[i]; av[i + 8] = a2[i]; bv[i] = b1[i]; bv[i + 8] = b2[i]; }
    c = __builtin_amdgcn_wmma_f32_16x16x32_f16(false, av, false, bv, (short)0, c, false, false);
  }
  float bz = bias ? bias[nt * 16 + l16] : 0.f;
#pragma unroll
  for (int r = 0; r < 8; ++r) {
    int m = mt * 16 + r + 8 * hi;
    if (m < M) C[(long long)m * N + nt * 16 + l16] = c[r] + bz;
  }
}

// ---------------- GRU sequential scan (one block per batch elem) ----------------
// whh [3E,E] cached in LDS (196KB, legal on CDNA5: 320KB/WG). 384 threads: one per gate row.
// Next-step gx row is prefetched (global_prefetch_b8) behind the LDS matvec.
__global__ void gru_scan(const float* __restrict__ gx, const float* __restrict__ whh,
                         const float* __restrict__ bhh, const float* __restrict__ h0,
                         float* __restrict__ out, float* __restrict__ hT,
                         int reverse, int outOff, int outStride) {
  extern __shared__ float sm[];
  float* whh_s = sm;                 // 384*128
  float* h_s   = sm + H3 * EE;       // 128
  float* gh_s  = h_s + EE;           // 384
  int b = blockIdx.x, t = threadIdx.x;
  for (int i = t; i < H3 * EE; i += H3) whh_s[i] = whh[i];
  if (t < EE) h_s[t] = h0 ? h0[b * EE + t] : 0.f;
  float bh = bhh[t];
  __syncthreads();
  for (int s = 0; s < SS; ++s) {
    int sa = reverse ? (SS - 1 - s) : s;
    // prefetch next step's gate inputs while we do the LDS matvec
    if (s + 1 < SS) {
      int sn = reverse ? (SS - 2 - s) : (s + 1);
      __builtin_prefetch(gx + (long long)(b * SS + sn) * H3 + t, 0, 0);
    }
    const float* wrow = whh_s + t * EE;
    float acc = bh;
#pragma unroll 8
    for (int e = 0; e < EE; ++e) acc += wrow[e] * h_s[e];
    gh_s[t] = acc;
    __syncthreads();
    float h2 = 0.f;
    if (t < EE) {
      const float* gxr = gx + (long long)(b * SS + sa) * H3;
      float r = sigmf(gxr[t] + gh_s[t]);
      float z = sigmf(gxr[EE + t] + gh_s[EE + t]);
      float n = tanhf(gxr[2 * EE + t] + r * gh_s[2 * EE + t]);
      h2 = (1.f - z) * n + z * h_s[t];
      out[(long long)(b * SS + sa) * outStride + outOff + t] = h2;
    }
    __syncthreads();
    if (t < EE) h_s[t] = h2;
    __syncthreads();
  }
  if (hT && t < EE) hT[b * EE + t] = h_s[t];
}

// ---------------- factored additive scores: sc[b,q,k] = sum_e v[e]*tanh(P[b,k,e] + sgn*Q[b,q,e]) ----
__global__ void score_af(const float* __restrict__ P, const float* __restrict__ Q,
                         const float* __restrict__ v, float* __restrict__ sc, float sgn) {
  __shared__ float qv[EE], vv[EE];
  int bq = blockIdx.x, b = bq >> 8, t = threadIdx.x;
  if (t < EE) { qv[t] = Q[(long long)bq * EE + t]; vv[t] = v[t]; }
  __syncthreads();
  const float* prow = P + (long long)(b * SS + t) * EE;
  float acc = 0.f;
  for (int e = 0; e < EE; ++e) acc += vv[e] * tanhf(prow[e] + sgn * qv[e]);
  sc[(long long)bq * SS + t] = acc;
}

// ---------------- fused mul-project score (ptd/pts): sc[b,q,k] = v . tanh(W (X_k ⊙ Y_q)) -------
// one wave per (q, 16-wide k-tile); elementwise f16 product fused into A-fragment load.
__global__ void mulproj_score(const _Float16* __restrict__ X, const _Float16* __restrict__ Y,
                              const _Float16* __restrict__ W, const float* __restrict__ v,
                              float* __restrict__ sc) {
  int bq = blockIdx.y, b = bq >> 8;
  int kt = blockIdx.x;
  int lane = threadIdx.x;
  int hi = lane >> 4, l16 = lane & 15;
  const _Float16* yrow = Y + (long long)bq * E2;
  const _Float16* xrow = X + (long long)(b * SS + kt * 16 + l16) * E2;
  int r1 = hi * 8, r2 = 16 + hi * 8;
  float scacc[8] = {0.f};
  for (int et = 0; et < 8; ++et) {
    v8f c = {};
    const _Float16* wrow = W + (long long)(et * 16 + l16) * E2;
#pragma unroll
    for (int k = 0; k < E2; k += 32) {
      half8 xa = *(const half8*)(xrow + k + r1);
      half8 xb = *(const half8*)(xrow + k + r2);
      half8 ya = *(const half8*)(yrow + k + r1);
      half8 yb = *(const half8*)(yrow + k + r2);
      half8 a1 = xa * ya, a2 = xb * yb;          // v_pk_mul_f16
      half8 b1 = *(const half8*)(wrow + k + r1);
      half8 b2 = *(const half8*)(wrow + k + r2);
      v16h av, bv;
#pragma unroll
      for (int i = 0; i < 8; ++i) { av[i] = a1[i]; av[i + 8] = a2[i]; bv[i] = b1[i]; bv[i + 8] = b2[i]; }
      c = __builtin_amdgcn_wmma_f32_16x16x32_f16(false, av, false, bv, (short)0, c, false, false);
    }
    float vn = v[et * 16 + l16];
#pragma unroll
    for (int r = 0; r < 8; ++r) {
      float val = vn * tanhf(c[r]);
#pragma unroll
      for (int o = 8; o; o >>= 1) val += __shfl_xor(val, o, 16);
      scacc[r] += val;
    }
  }
  if (l16 == 0) {
#pragma unroll
    for (int r = 0; r < 8; ++r)
      sc[(long long)bq * SS + kt * 16 + r + 8 * hi] = scacc[r];
  }
}

// ---------------- softmax over K=256 + weighted sum of V rows (Dv==blockDim==256) ----------------
__global__ void attn_apply(const float* __restrict__ sc, const float* __restrict__ V,
                           float* __restrict__ out, int Dv, int qPerB) {
  __shared__ float wsm[SS];
  __shared__ float red[8];
  int g = blockIdx.x, t = threadIdx.x;
  int b = g / qPerB;
  float x = sc[(long long)g * SS + t];
  float m = x;
  for (int o = 16; o; o >>= 1) m = fmaxf(m, __shfl_xor(m, o, 32));
  if ((t & 31) == 0) red[t >> 5] = m;
  __syncthreads();
  if (t < 8) { float mm = red[t]; for (int o = 4; o; o >>= 1) mm = fmaxf(mm, __shfl_xor(mm, o, 8)); red[t] = mm; }
  __syncthreads();
  m = red[0];
  float e = __expf(x - m);
  float s = e;
  for (int o = 16; o; o >>= 1) s += __shfl_xor(s, o, 32);
  __syncthreads();
  if ((t & 31) == 0) red[t >> 5] = s;
  __syncthreads();
  if (t < 8) { float ss = red[t]; for (int o = 4; o; o >>= 1) ss += __shfl_xor(ss, o, 8); red[t] = ss; }
  __syncthreads();
  s = red[0];
  wsm[t] = e / s;
  __syncthreads();
  const float* vb = V + (long long)(b * SS) * Dv;
  float acc = 0.f;
  for (int k = 0; k < SS; ++k) acc += wsm[k] * vb[(long long)k * Dv + t];
  out[(long long)g * Dv + t] = acc;
}

// ---------------- row score: out[r] = sum_e v[e]*tanh(T[r,e]) ----------------
__global__ void rowscore(const float* __restrict__ T, const float* __restrict__ v,
                         float* __restrict__ out, int R) {
  int r = blockIdx.x * blockDim.x + threadIdx.x;
  if (r >= R) return;
  const float* row = T + (long long)r * EE;
  float acc = 0.f;
  for (int e = 0; e < EE; ++e) acc += v[e] * tanhf(row[e]);
  out[r] = acc;
}

// ---------------- final pooling score (no tanh): sc[b,s] = v . (A1[b,s,:] + A2[b,:]) ----------------
__global__ void finalscore(const float* __restrict__ A1, const float* __restrict__ A2,
                           const float* __restrict__ v, float* __restrict__ out) {
  __shared__ float a2[EE], vv[EE];
  int b = blockIdx.x, t = threadIdx.x;
  if (t < EE) { a2[t] = A2[b * EE + t]; vv[t] = v[t]; }
  __syncthreads();
  const float* row = A1 + (long long)(b * SS + t) * EE;
  float acc = 0.f;
  for (int e = 0; e < EE; ++e) acc += vv[e] * (row[e] + a2[e]);
  out[b * SS + t] = acc;
}

// ---------------- concat 6 x [B,S,2E] -> [B,S,12E] ----------------
__global__ void concat6(const float* hq, const float* pts_, const float* ptc_,
                        const float* ptd_, const float* ptb_, const float* ptm_,
                        float* __restrict__ agg) {
  int bs = blockIdx.x, d = threadIdx.x;
  long long i = (long long)bs * E2 + d;
  float* row = agg + (long long)bs * E12;
  row[d] = hq[i]; row[E2 + d] = pts_[i]; row[2 * E2 + d] = ptc_[i];
  row[3 * E2 + d] = ptd_[i]; row[4 * E2 + d] = ptb_[i]; row[5 * E2 + d] = ptm_[i];
}

// ---------------- prediction: sigmoid(rr @ Wpred^T) ----------------
__global__ void pred_kernel(const float* __restrict__ rr, const float* __restrict__ Wp,
                            float* __restrict__ out) {
  int i = threadIdx.x;
  if (i >= BB * LL) return;
  int b = i / LL, l = i % LL;
  float acc = 0.f;
  for (int d = 0; d < E2; ++d) acc += rr[b * E2 + d] * Wp[l * E2 + d];
  out[i] = sigmf(acc);
}

// ================= host side =================
static void gemm(const _Float16* A, const _Float16* W, const float* bias, float* C,
                 int M, int N, int Kp, int batch, long long sA, long long sW, long long sC,
                 hipStream_t st) {
  dim3 g(N / 16, (M + 15) / 16, batch);
  wmma_gemm<<<g, dim3(32), 0, st>>>(A, W, bias, C, M, N, Kp, sA, sW, sC);
}
static void cvt(const float* in, _Float16* out, int Rv, int R, int K, int Kp, hipStream_t st) {
  long long n = (long long)R * Kp;
  cvt_f16_pad<<<(int)((n + 255) / 256), 256, 0, st>>>(in, out, Rv, R, K, Kp);
}
static void scan(const float* gx, const float* whh, const float* bhh, const float* h0,
                 float* out, float* hT, int rev, int off, hipStream_t st) {
  size_t shm = (size_t)(H3 * EE + EE + H3) * sizeof(float);
  gru_scan<<<BB, H3, shm, st>>>(gx, whh, bhh, h0, out, hT, rev, off, E2);
}

extern "C" void kernel_launch(void* const* d_in, const int* in_sizes, int n_in,
                              void* d_out, int out_size, void* d_ws, size_t ws_size,
                              hipStream_t stream) {
  const int*   inputs   = (const int*)d_in[0];
  const float* emb      = (const float*)d_in[1];
  const float* e_wih_f  = (const float*)d_in[2];
  const float* e_whh_f  = (const float*)d_in[3];
  const float* e_bih_f  = (const float*)d_in[4];
  const float* e_bhh_f  = (const float*)d_in[5];
  const float* e_wih_b  = (const float*)d_in[6];
  const float* e_whh_b  = (const float*)d_in[7];
  const float* e_bih_b  = (const float*)d_in[8];
  const float* e_bhh_b  = (const float*)d_in[9];
  const float* h_wih_f  = (const float*)d_in[10];
  const float* h_whh_f  = (const float*)d_in[11];
  const float* h_bih_f  = (const float*)d_in[12];
  const float* h_bhh_f  = (const float*)d_in[13];
  const float* h_wih_b  = (const float*)d_in[14];
  const float* h_whh_b  = (const float*)d_in[15];
  const float* h_bih_b  = (const float*)d_in[16];
  const float* h_bhh_b  = (const float*)d_in[17];
  const float* a_wih_f  = (const float*)d_in[18];
  const float* a_whh_f  = (const float*)d_in[19];
  const float* a_bih_f  = (const float*)d_in[20];
  const float* a_bhh_f  = (const float*)d_in[21];
  const float* a_wih_b  = (const float*)d_in[22];
  const float* a_whh_b  = (const float*)d_in[23];
  const float* a_bih_b  = (const float*)d_in[24];
  const float* a_bhh_b  = (const float*)d_in[25];
  const float* Wc1 = (const float*)d_in[26];
  const float* Wc2 = (const float*)d_in[27];
  const float* vc  = (const float*)d_in[28];
  const float* Wb  = (const float*)d_in[29];
  const float* Wd  = (const float*)d_in[30];
  const float* vd  = (const float*)d_in[31];
  const float* Wm  = (const float*)d_in[32];
  const float* vm  = (const float*)d_in[33];
  const float* Ws  = (const float*)d_in[34];
  const float* vs  = (const float*)d_in[35];
  const float* Wp  = (const float*)d_in[36];
  const float* vp  = (const float*)d_in[37];
  const float* Wpred = (const float*)d_in[38];

  // -------- workspace carve-out --------
  char* base = (char*)d_ws;
  size_t off = 0;
  auto alloc = [&](size_t bytes) -> void* {
    off = (off + 255) & ~(size_t)255;
    void* p = base + off; off += bytes; return p;
  };
  const int R = BB * SS;                                   // 1024 rows
  float*     x       = (float*)alloc((size_t)R * DD * 4);
  _Float16*  x16     = (_Float16*)alloc((size_t)R * DP * 2);
  _Float16*  wEf16   = (_Float16*)alloc((size_t)H3 * DP * 2);
  _Float16*  wEb16   = (_Float16*)alloc((size_t)H3 * DP * 2);
  float*     gxf     = (float*)alloc((size_t)R * H3 * 4);
  float*     gxb     = (float*)alloc((size_t)R * H3 * 4);
  float*     hp      = (float*)alloc((size_t)R * E2 * 4);
  float*     hq      = (float*)alloc((size_t)R * E2 * 4);
  float*     aggrep  = (float*)alloc((size_t)R * E2 * 4);
  float*     hTf     = (float*)alloc((size_t)BB * EE * 4);
  float*     hTb     = (float*)alloc((size_t)BB * EE * 4);
  _Float16*  hp16    = (_Float16*)alloc((size_t)R * E2 * 2);
  _Float16*  hq16    = (_Float16*)alloc((size_t)R * E2 * 2);
  _Float16*  ar16    = (_Float16*)alloc((size_t)R * E2 * 2);
  _Float16*  wHf16   = (_Float16*)alloc((size_t)H3 * E2 * 2);
  _Float16*  wHb16   = (_Float16*)alloc((size_t)H3 * E2 * 2);
  _Float16*  Wc1_16  = (_Float16*)alloc((size_t)EE * E2 * 2);
  _Float16*  Wc2_16  = (_Float16*)alloc((size_t)EE * E2 * 2);
  _Float16*  Wd16    = (_Float16*)alloc((size_t)EE * E2 * 2);
  _Float16*  Wm16    = (_Float16*)alloc((size_t)EE * E2 * 2);
  _Float16*  Ws16    = (_Float16*)alloc((size_t)EE * E2 * 2);
  _Float16*  Wp16    = (_Float16*)alloc((size_t)EE * E2 * 2);
  _Float16*  Wb16    = (_Float16*)alloc((size_t)E2 * E2 * 2);
  float*     e1      = (float*)alloc((size_t)R * EE * 4);  // s1 / Pm / Tp / A1
  float*     e2      = (float*)alloc((size_t)R * EE * 4);  // s2 / Qm
  float*     tbuf    = (float*)alloc((size_t)R * E2 * 4);  // hp @ Wb^T
  _Float16*  t16     = (_Float16*)alloc((size_t)R * E2 * 2);
  float*     scb     = (float*)alloc((size_t)R * SS * 4);  // [B,S,S] scores (reused)
  float*     ptc     = (float*)alloc((size_t)R * E2 * 4);
  float*     ptb     = (float*)alloc((size_t)R * E2 * 4);
  float*     ptd     = (float*)alloc((size_t)R * E2 * 4);
  float*     ptm     = (float*)alloc((size_t)R * E2 * 4);
  float*     pts     = (float*)alloc((size_t)R * E2 * 4);
  float*     agg     = (float*)alloc((size_t)R * E12 * 4);
  _Float16*  agg16   = (_Float16*)alloc((size_t)R * E12 * 2);
  _Float16*  wAf16   = (_Float16*)alloc((size_t)H3 * E12 * 2);
  _Float16*  wAb16   = (_Float16*)alloc((size_t)H3 * E12 * 2);
  float*     sj      = (float*)alloc((size_t)R * 4);
  float*     rl      = (float*)alloc((size_t)BB * E2 * 4);
  _Float16*  rl16    = (_Float16*)alloc((size_t)16 * E2 * 2);  // row-padded to 16
  float*     A2b     = (float*)alloc((size_t)BB * EE * 4);
  float*     scfin   = (float*)alloc((size_t)BB * SS * 4);
  float*     rr      = (float*)alloc((size_t)BB * E2 * 4);
  (void)ws_size; (void)in_sizes; (void)n_in; (void)out_size;

  // -------- 1) embedding --------
  embed_gather<<<R, 128, 0, stream>>>(inputs, emb, x);
  cvt(x, x16, R, R, DD, DP, stream);

  // -------- 2) encoder biGRU --------
  cvt(e_wih_f, wEf16, H3, H3, DD, DP, stream);
  cvt(e_wih_b, wEb16, H3, H3, DD, DP, stream);
  gemm(x16, wEf16, e_bih_f, gxf, R, H3, DP, 1, 0, 0, 0, stream);
  gemm(x16, wEb16, e_bih_b, gxb, R, H3, DP, 1, 0, 0, 0, stream);
  scan(gxf, e_whh_f, e_bhh_f, nullptr, hp, hTf, 0, 0, stream);
  scan(gxb, e_whh_b, e_bhh_b, nullptr, hp, hTb, 1, EE, stream);
  cvt(hp, hp16, R, R, E2, E2, stream);

  // -------- 3) hidden biGRU (h0 = encoder final states) --------
  cvt(h_wih_f, wHf16, H3, H3, E2, E2, stream);
  cvt(h_wih_b, wHb16, H3, H3, E2, E2, stream);
  gemm(hp16, wHf16, h_bih_f, gxf, R, H3, E2, 1, 0, 0, 0, stream);
  gemm(hp16, wHb16, h_bih_b, gxb, R, H3, E2, 1, 0, 0, 0, stream);
  scan(gxf, h_whh_f, h_bhh_f, hTf, hq, nullptr, 0, 0, stream);
  scan(gxb, h_whh_b, h_bhh_b, hTb, hq, nullptr, 1, EE, stream);
  cvt(hq, hq16, R, R, E2, E2, stream);

  // -------- 4) attentions --------
  // ptc: additive, factored
  cvt(Wc1, Wc1_16, EE, EE, E2, E2, stream);
  cvt(Wc2, Wc2_16, EE, EE, E2, E2, stream);
  gemm(hp16, Wc1_16, nullptr, e1, R, EE, E2, 1, 0, 0, 0, stream);
  gemm(hq16, Wc2_16, nullptr, e2, R, EE, E2, 1, 0, 0, 0, stream);
  score_af<<<R, SS, 0, stream>>>(e1, e2, vc, scb, 1.f);
  attn_apply<<<R, SS, 0, stream>>>(scb, hp, ptc, E2, SS);
  // ptb: bilinear (batched WMMA GEMM for scores)
  cvt(Wb, Wb16, E2, E2, E2, E2, stream);
  gemm(hp16, Wb16, nullptr, tbuf, R, E2, E2, 1, 0, 0, 0, stream);
  cvt(tbuf, t16, R, R, E2, E2, stream);
  gemm(hq16, t16, nullptr, scb, SS, SS, E2, BB,
       (long long)SS * E2, (long long)SS * E2, (long long)SS * SS, stream);
  attn_apply<<<R, SS, 0, stream>>>(scb, hp, ptb, E2, SS);
  // ptd: fused mul-project WMMA
  cvt(Wd, Wd16, EE, EE, E2, E2, stream);
  mulproj_score<<<dim3(SS / 16, R), 32, 0, stream>>>(hp16, hq16, Wd16, vd, scb);
  attn_apply<<<R, SS, 0, stream>>>(scb, hp, ptd, E2, SS);
  // ptm: difference, factored (Wm(hp-hq) = Wm hp - Wm hq)
  cvt(Wm, Wm16, EE, EE, E2, E2, stream);
  gemm(hp16, Wm16, nullptr, e1, R, EE, E2, 1, 0, 0, 0, stream);
  gemm(hq16, Wm16, nullptr, e2, R, EE, E2, 1, 0, 0, 0, stream);
  score_af<<<R, SS, 0, stream>>>(e1, e2, vm, scb, -1.f);
  attn_apply<<<R, SS, 0, stream>>>(scb, hp, ptm, E2, SS);
  // pts: self mul-project
  cvt(Ws, Ws16, EE, EE, E2, E2, stream);
  mulproj_score<<<dim3(SS / 16, R), 32, 0, stream>>>(hq16, hq16, Ws16, vs, scb);
  attn_apply<<<R, SS, 0, stream>>>(scb, hq, pts, E2, SS);

  // -------- 5) aggregation biGRU --------
  concat6<<<R, E2, 0, stream>>>(hq, pts, ptc, ptd, ptb, ptm, agg);
  cvt(agg, agg16, R, R, E12, E12, stream);
  cvt(a_wih_f, wAf16, H3, H3, E12, E12, stream);
  cvt(a_wih_b, wAb16, H3, H3, E12, E12, stream);
  gemm(agg16, wAf16, a_bih_f, gxf, R, H3, E12, 1, 0, 0, 0, stream);
  gemm(agg16, wAb16, a_bih_b, gxb, R, H3, E12, 1, 0, 0, 0, stream);
  scan(gxf, a_whh_f, a_bhh_f, nullptr, aggrep, nullptr, 0, 0, stream);
  scan(gxb, a_whh_b, a_bhh_b, nullptr, aggrep, nullptr, 1, EE, stream);

  // -------- 6) pooled query rl over hq --------
  cvt(Wp, Wp16, EE, EE, E2, E2, stream);
  gemm(hq16, Wp16, nullptr, e1, R, EE, E2, 1, 0, 0, 0, stream);   // Tp
  rowscore<<<(R + 255) / 256, 256, 0, stream>>>(e1, vp, sj, R);
  attn_apply<<<BB, SS, 0, stream>>>(sj, hq, rl, E2, 1);

  // -------- 7) final pooling over agg_rep --------
  cvt(aggrep, ar16, R, R, E2, E2, stream);
  cvt(rl, rl16, BB, 16, E2, E2, stream);                          // rows 4..15 zeroed
  gemm(ar16, Wc1_16, nullptr, e1, R, EE, E2, 1, 0, 0, 0, stream); // A1
  gemm(rl16, Wc2_16, nullptr, A2b, BB, EE, E2, 1, 0, 0, 0, stream);
  finalscore<<<BB, SS, 0, stream>>>(e1, A2b, vc, scfin);
  attn_apply<<<BB, SS, 0, stream>>>(scfin, aggrep, rr, E2, 1);

  // -------- 8) prediction --------
  pred_kernel<<<1, 128, 0, stream>>>(rr, Wpred, (float*)d_out);
}
